// SwinTreeTransformerBlock_64115271794698
// MI455X (gfx1250) — matrix operand order repaired and checked by reading, not compiled, not run on hardware
//
#include <hip/hip_runtime.h>
#include <hip/hip_bf16.h>
#include <cstdint>

// Problem constants (from reference setup_inputs)
#define WSZ    7
#define NHEAD  12
#define SEQ    448
#define CH     768
#define HID    3072
#define C3     2304
#define NWIN   64              // SEQ / WSZ
#define BATCH  64
#define MROWS  (BATCH * SEQ)   // 28672
#define NATTNW (BATCH * NWIN * NHEAD)  // 49152 waves

typedef __attribute__((ext_vector_type(16))) __bf16 v16bf;
typedef __attribute__((ext_vector_type(8)))  float  v8f;

union FragBF {
  struct { uint4 lo, hi; } u;
  v16bf v;
};

__device__ __forceinline__ unsigned short f2bf(float f) {
  unsigned u = __builtin_bit_cast(unsigned, f);
  unsigned r = u + 0x7fffu + ((u >> 16) & 1u);          // round-to-nearest-even
  if ((u & 0x7f800000u) == 0x7f800000u) r = u;          // inf/nan: truncate
  return (unsigned short)(r >> 16);
}

__device__ __forceinline__ float bf2f(unsigned short h) {
  unsigned u = (unsigned)h << 16;
  return __builtin_bit_cast(float, u);
}

// -------------------------------------------------------------------------
// f32 -> bf16 bulk convert (weights)
// -------------------------------------------------------------------------
__global__ __launch_bounds__(256) void cvt_bf16_kernel(
    const float* __restrict__ in, unsigned short* __restrict__ out, int n) {
  int i = (blockIdx.x * 256 + threadIdx.x) * 4;
  if (i + 3 < n) {
    float4 f = *(const float4*)(in + i);
    unsigned short r[4] = {f2bf(f.x), f2bf(f.y), f2bf(f.z), f2bf(f.w)};
    *(uint2*)(out + i) = *(const uint2*)r;
  }
}

// -------------------------------------------------------------------------
// LayerNorm (f32 in) -> bf16 out, with roll(-shift) fused into the write.
// One block per row; 256 threads; C=768 -> 3 elems/thread.
// -------------------------------------------------------------------------
__global__ __launch_bounds__(256) void ln_bf16_kernel(
    const float* __restrict__ x, const float* __restrict__ g,
    const float* __restrict__ bta, unsigned short* __restrict__ out, int shift) {
  const int row = blockIdx.x;
  const float* xr = x + (size_t)row * CH;
  float v[3], s1 = 0.f, s2 = 0.f;
  for (int k = 0; k < 3; ++k) {
    v[k] = xr[threadIdx.x + k * 256];
    s1 += v[k];
    s2 += v[k] * v[k];
  }
  for (int off = 16; off; off >>= 1) {
    s1 += __shfl_xor(s1, off, 32);
    s2 += __shfl_xor(s2, off, 32);
  }
  __shared__ float r1[8], r2[8];
  const int wv = threadIdx.x >> 5;
  if ((threadIdx.x & 31) == 0) { r1[wv] = s1; r2[wv] = s2; }
  __syncthreads();
  float t1 = 0.f, t2 = 0.f;
  for (int i = 0; i < 8; ++i) { t1 += r1[i]; t2 += r2[i]; }
  const float mean = t1 * (1.0f / CH);
  const float var  = t2 * (1.0f / CH) - mean * mean;
  const float rstd = rsqrtf(var + 1e-5f);

  const int b = row / SEQ, l = row % SEQ;
  int lo = l - shift; if (lo < 0) lo += SEQ;             // xs[lo] = xn[l]
  unsigned short* orow = out + (size_t)(b * SEQ + lo) * CH;
  for (int k = 0; k < 3; ++k) {
    const int i = threadIdx.x + k * 256;
    orow[i] = f2bf((v[k] - mean) * rstd * g[i] + bta[i]);
  }
}

// -------------------------------------------------------------------------
// Tiled bf16 WMMA GEMM:  C[M,N] = A[M,K] @ W[N,K]^T + bias
// EPI 0: f32 out.  EPI 1: exact-GELU, bf16 out.  EPI 2: plain bf16 out.
// Block tile 128x128, K-step 32, 8 waves of 64x32 (4x2 WMMA frags).
// Double-buffered LDS fed by global_load_async_to_lds_b128 (ASYNCcnt),
// so the copy of tile t+1 overlaps the WMMAs of tile t.
// Requires M%128==0, N%128==0, K%32==0 (true for all call sites).
// -------------------------------------------------------------------------
template<int EPI>
__global__ __launch_bounds__(256) void gemm_bf16_wmma(
    const unsigned short* __restrict__ A, const unsigned short* __restrict__ Bw,
    const float* __restrict__ bias, void* __restrict__ Cout,
    int M, int N, int K) {
  // 2 buffers x (A tile 128x40h + B tile 128x40h) = 40960 B
  __shared__ unsigned short pool[2 * 2 * 128 * 40];
  // Low 32 bits of an LDS flat address are the raw LDS byte offset (ISA 10.2).
  const unsigned lds0 = (unsigned)(uintptr_t)(void*)pool;

  const int tid  = threadIdx.x;
  const int m0   = blockIdx.y * 128;
  const int n0   = blockIdx.x * 128;
  const int wave = tid >> 5;
  const int lane = tid & 31;
  const int half = lane >> 4;     // 0 | 1
  const int lr   = lane & 15;
  const int wm   = wave >> 2;     // 0..1 -> 64-row slab
  const int wn   = wave & 3;      // 0..3 -> 32-col slab

  v8f acc[4][2] = {};

  const int kc = (tid & 3) * 8;            // 8-half chunk within 32-wide K slice
  const int r0 = tid >> 2;                 // rows r0 and r0+64
  const unsigned coff0 = (unsigned)(r0 * 80 + kc * 2);          // 16B aligned
  const unsigned coff1 = (unsigned)((r0 + 64) * 80 + kc * 2);

  // Issue the 4 per-thread async 16B copies for K-tile at `kk` into buffer p.
  auto issue = [&](int p, int kk) {
    const unsigned ab = lds0 + (unsigned)p * 20480u;
    const unsigned bb = ab + 10240u;
    const unsigned short* pa0 = A  + (size_t)(m0 + r0) * K + kk + kc;
    const unsigned short* pa1 = pa0 + (size_t)64 * K;
    const unsigned short* pb0 = Bw + (size_t)(n0 + r0) * K + kk + kc;
    const unsigned short* pb1 = pb0 + (size_t)64 * K;
    unsigned long long gA0 = (unsigned long long)(uintptr_t)pa0;
    unsigned long long gA1 = (unsigned long long)(uintptr_t)pa1;
    unsigned long long gB0 = (unsigned long long)(uintptr_t)pb0;
    unsigned long long gB1 = (unsigned long long)(uintptr_t)pb1;
    asm volatile("global_load_async_to_lds_b128 %0, %1, off"
                 :: "v"(ab + coff0), "v"(gA0) : "memory");
    asm volatile("global_load_async_to_lds_b128 %0, %1, off"
                 :: "v"(ab + coff1), "v"(gA1) : "memory");
    asm volatile("global_load_async_to_lds_b128 %0, %1, off"
                 :: "v"(bb + coff0), "v"(gB0) : "memory");
    asm volatile("global_load_async_to_lds_b128 %0, %1, off"
                 :: "v"(bb + coff1), "v"(gB1) : "memory");
    if (kk + 32 < K) {                     // HBM->L2 one more tile ahead
      __builtin_prefetch(pa0 + 32, 0, 1);
      __builtin_prefetch(pb0 + 32, 0, 1);
    }
  };

  const int nt = K / 32;
  issue(0, 0);
  for (int t = 0; t < nt; ++t) {
    asm volatile("s_wait_asynccnt 0x0" ::: "memory");   // tile t resident
    __syncthreads();                                    // ..in every wave
    if (t + 1 < nt) issue((t + 1) & 1, (t + 1) * 32);   // overlaps compute(t)

    const unsigned short* Ap = pool + (size_t)(t & 1) * 10240;
    const unsigned short* Bp = Ap + 5120;

    FragBF a[4], b[2];
    for (int i = 0; i < 4; ++i) {
      const int ar = (wm * 64 + i * 16 + lr) * 40;
      a[i].u.lo = *(const uint4*)&Ap[ar + half * 8];        // K kb..kb+7
      a[i].u.hi = *(const uint4*)&Ap[ar + 16 + half * 8];   // K kb+16..kb+23
    }
    for (int j = 0; j < 2; ++j) {
      const int br = (wn * 32 + j * 16 + lr) * 40;
      b[j].u.lo = *(const uint4*)&Bp[br + half * 8];        // K kb..kb+7
      b[j].u.hi = *(const uint4*)&Bp[br + 16 + half * 8];   // K kb+16..kb+23
    }
    for (int i = 0; i < 4; ++i)
      for (int j = 0; j < 2; ++j)
        acc[i][j] = __builtin_amdgcn_wmma_f32_16x16x32_bf16(
            false, a[i].v, false, b[j].v, (short)0, acc[i][j], false, false);
  }

  // ---- epilogue: f32 C layout (VGPR r -> M = 8*half + r, N = lr) ----
  for (int i = 0; i < 4; ++i) {
    const int mbase = m0 + wm * 64 + i * 16 + half * 8;
    for (int j = 0; j < 2; ++j) {
      const int ncol = n0 + wn * 32 + j * 16 + lr;
      const float bv = bias[ncol];
      for (int r = 0; r < 8; ++r) {
        float v = acc[i][j][r] + bv;
        if (EPI == 1) {
          v = 0.5f * v * (1.0f + erff(v * 0.70710678118654752f));
          ((unsigned short*)Cout)[(size_t)(mbase + r) * N + ncol] = f2bf(v);
        } else if (EPI == 2) {
          ((unsigned short*)Cout)[(size_t)(mbase + r) * N + ncol] = f2bf(v);
        } else {
          ((float*)Cout)[(size_t)(mbase + r) * N + ncol] = v;
        }
      }
    }
  }
}

// -------------------------------------------------------------------------
// Windowed attention: one wave per (window, head). N=7, hd=64 (2 f32/lane).
// qkv: bf16 [MROWS, 2304] in (rolled) window order. out: bf16 [MROWS, 768].
// Mask codes: plain -> {0,1}, allowed iff ci==cj && ci!=0 (== both nonzero);
// shifted -> {0,1,2} replicating the minf-diff semantics incl. NaN rows.
// Fully-masked rows produce 0 (nan_to_num).
// -------------------------------------------------------------------------
__global__ __launch_bounds__(256) void win_attn_kernel(
    const unsigned short* __restrict__ qkv, const int* __restrict__ xmask,
    const float* __restrict__ gprob, unsigned short* __restrict__ out, int shift) {
  const int wid  = blockIdx.x * 8 + (threadIdx.x >> 5);   // 0..NATTNW-1
  const int lane = threadIdx.x & 31;
  const int win  = wid / NHEAD;
  const int h    = wid % NHEAD;
  const int b    = win / NWIN;
  const int w    = win % NWIN;

  float qf[WSZ][2], kf[WSZ][2], vf[WSZ][2];
  for (int j = 0; j < WSZ; ++j) {
    const size_t base = (size_t)(win * WSZ + j) * C3 + h * 64;
    qf[j][0] = bf2f(qkv[base + lane])      * 0.125f;   // scale = hd^-0.5
    qf[j][1] = bf2f(qkv[base + 32 + lane]) * 0.125f;
    kf[j][0] = bf2f(qkv[base + CH + lane]);
    kf[j][1] = bf2f(qkv[base + CH + 32 + lane]);
    vf[j][0] = bf2f(qkv[base + 2 * CH + lane]);
    vf[j][1] = bf2f(qkv[base + 2 * CH + 32 + lane]);
  }

  int code[WSZ], lorig[WSZ];
  for (int j = 0; j < WSZ; ++j) {
    int lr = w * WSZ + j;
    int lo = lr + shift; if (lo >= SEQ) lo -= SEQ;        // original position
    lorig[j] = lo;
    const int mv = xmask[b * SEQ + lo];
    code[j] = shift ? ((lo < shift ? 1 : 2) * mv) : mv;
  }

  for (int i = 0; i < WSZ; ++i) {
    float s[WSZ];
    for (int j = 0; j < WSZ; ++j) {
      float p = qf[i][0] * kf[j][0] + qf[i][1] * kf[j][1];
      for (int off = 16; off; off >>= 1) p += __shfl_xor(p, off, 32);
      s[j] = p;
    }
    float mx = -3.4e38f;
    bool any = false;
    for (int j = 0; j < WSZ; ++j) {
      const bool al = (code[i] == code[j]) && (code[i] != 0);
      if (al) { any = true; mx = fmaxf(mx, s[j]); }
    }
    float pj[WSZ], sum = 0.f;
    for (int j = 0; j < WSZ; ++j) {
      const bool al = (code[i] == code[j]) && (code[i] != 0);
      const float e = al ? __expf(s[j] - mx) : 0.0f;
      pj[j] = e; sum += e;
    }
    const float inv = any ? (1.0f / sum) : 0.0f;

    const size_t gbase = ((size_t)b * SEQ + lorig[i]) * SEQ;
    float a0 = 0.f, a1 = 0.f;
    for (int j = 0; j < WSZ; ++j) {
      const float pv = pj[j] * inv * gprob[gbase + lorig[j]];
      a0 += pv * vf[j][0];
      a1 += pv * vf[j][1];
    }
    const size_t ob = (size_t)(win * WSZ + i) * CH + h * 64;
    out[ob + lane]      = f2bf(a0);
    out[ob + 32 + lane] = f2bf(a1);
  }
}

// -------------------------------------------------------------------------
// out[b,l,:] = shortcut[b,l,:] + delta[b,(l-shift) mod L,:]
// (roll(+shift) of the rolled-space GEMM output fused into the read)
// -------------------------------------------------------------------------
__global__ __launch_bounds__(256) void residual_shift_kernel(
    const float* __restrict__ sc, const float* __restrict__ delta,
    float* __restrict__ out, int shift) {
  const int row = blockIdx.x;
  const int b = row / SEQ, l = row % SEQ;
  int ls = l - shift; if (ls < 0) ls += SEQ;
  const float* d = delta + (size_t)(b * SEQ + ls) * CH;
  const float* s = sc + (size_t)row * CH;
  float* o = out + (size_t)row * CH;
  for (int k = 0; k < 3; ++k) {
    const int i = threadIdx.x + k * 256;
    o[i] = s[i] + d[i];
  }
}

// -------------------------------------------------------------------------
// Host orchestration
// -------------------------------------------------------------------------
extern "C" void kernel_launch(void* const* d_in, const int* in_sizes, int n_in,
                              void* d_out, int out_size, void* d_ws, size_t ws_size,
                              hipStream_t stream) {
  (void)in_sizes; (void)n_in; (void)out_size; (void)ws_size;
  const float* x      = (const float*)d_in[0];
  const int*   xmask  = (const int*)d_in[1];
  const float* gprob  = (const float*)d_in[2];
  const float* ln1_g  = (const float*)d_in[3];
  const float* ln1_b  = (const float*)d_in[4];
  const float* qkv_w  = (const float*)d_in[5];
  const float* qkv_b  = (const float*)d_in[6];
  const float* proj_w = (const float*)d_in[7];
  const float* proj_b = (const float*)d_in[8];
  const float* ln2_g  = (const float*)d_in[9];
  const float* ln2_b  = (const float*)d_in[10];
  const float* fc1_w  = (const float*)d_in[11];
  const float* fc1_b  = (const float*)d_in[12];
  const float* fc2_w  = (const float*)d_in[13];
  const float* fc2_b  = (const float*)d_in[14];

  char* ws = (char*)d_ws;
  size_t off = 0;
  auto take = [&](size_t bytes) -> char* {
    char* p = ws + off;
    off += (bytes + 255) & ~(size_t)255;
    return p;
  };
  unsigned short* w_qkv  = (unsigned short*)take((size_t)C3 * CH * 2);
  unsigned short* w_proj = (unsigned short*)take((size_t)CH * CH * 2);
  unsigned short* w_fc1  = (unsigned short*)take((size_t)HID * CH * 2);
  unsigned short* w_fc2  = (unsigned short*)take((size_t)CH * HID * 2);
  unsigned short* xln_bf = (unsigned short*)take((size_t)MROWS * CH * 2);
  unsigned short* h_bf   = (unsigned short*)take((size_t)MROWS * HID * 2);  // also qkv bf16
  unsigned short* att_bf = (unsigned short*)take((size_t)MROWS * CH * 2);
  float* big_f32 = (float*)take((size_t)MROWS * CH * 4);   // proj-out / fc2-out
  float* xmid    = (float*)take((size_t)MROWS * CH * 4);

  auto cvt = [&](const float* src, unsigned short* dst, size_t n) {
    cvt_bf16_kernel<<<dim3((unsigned)((n + 1023) / 1024)), 256, 0, stream>>>(src, dst, (int)n);
  };
  cvt(qkv_w,  w_qkv,  (size_t)C3 * CH);
  cvt(proj_w, w_proj, (size_t)CH * CH);
  cvt(fc1_w,  w_fc1,  (size_t)HID * CH);
  cvt(fc2_w,  w_fc2,  (size_t)CH * HID);

  auto sub_block = [&](const float* xin, float* xout, int shift) {
    ln_bf16_kernel<<<MROWS, 256, 0, stream>>>(xin, ln1_g, ln1_b, xln_bf, shift);
    gemm_bf16_wmma<2><<<dim3(C3 / 128, MROWS / 128), 256, 0, stream>>>(
        xln_bf, w_qkv, qkv_b, h_bf, MROWS, C3, CH);              // qkv as bf16
    win_attn_kernel<<<NATTNW / 8, 256, 0, stream>>>(h_bf, xmask, gprob, att_bf, shift);
    gemm_bf16_wmma<0><<<dim3(CH / 128, MROWS / 128), 256, 0, stream>>>(
        att_bf, w_proj, proj_b, big_f32, MROWS, CH, CH);
    residual_shift_kernel<<<MROWS, 256, 0, stream>>>(xin, big_f32, xmid, shift);
    ln_bf16_kernel<<<MROWS, 256, 0, stream>>>(xmid, ln2_g, ln2_b, xln_bf, 0);
    gemm_bf16_wmma<1><<<dim3(HID / 128, MROWS / 128), 256, 0, stream>>>(
        xln_bf, w_fc1, fc1_b, h_bf, MROWS, HID, CH);             // overwrites qkv
    gemm_bf16_wmma<0><<<dim3(CH / 128, MROWS / 128), 256, 0, stream>>>(
        h_bf, w_fc2, fc2_b, big_f32, MROWS, CH, HID);
    residual_shift_kernel<<<MROWS, 256, 0, stream>>>(xmid, big_f32, xout, 0);
  };

  float* out = (float*)d_out;
  sub_block(x,   out, 0);        // plain windows
  sub_block(out, out, WSZ / 2);  // shifted windows (shift = 3)
}
